// Model_39676907884409
// MI455X (gfx1250) — matrix operand order repaired and checked by reading, not compiled, not run on hardware
//
#include <hip/hip_runtime.h>
#include <math.h>

typedef __attribute__((ext_vector_type(16))) _Float16 v16h;
typedef __attribute__((ext_vector_type(8)))  _Float16 v8h;
typedef __attribute__((ext_vector_type(8)))  float    v8f;
typedef __attribute__((ext_vector_type(4)))  unsigned int u32x4;
typedef __attribute__((ext_vector_type(8)))  int i32x8;
typedef __attribute__((ext_vector_type(4)))  int i32x4;

#define NSLICE 512   // B*C = 64*8
#define SEQ    256
#define DHEAD  256

#if __has_builtin(__builtin_amdgcn_tensor_load_to_lds)
#define HAVE_TDM 1
#else
#define HAVE_TDM 0
#endif

static __device__ __forceinline__ v16h cat8(v8h lo, v8h hi) {
  return __builtin_shufflevector(lo, hi, 0,1,2,3,4,5,6,7,8,9,10,11,12,13,14,15);
}

#if HAVE_TDM
// Issue one 2D-tile TENSOR_LOAD_TO_LDS. Descriptor per CDNA5 ISA:
//  group0: [1:0]=count=1, [63:32]=lds byte addr, [120:64]=global byte addr, [127:126]=type=2
//  group1: [17:16]=data_size(1 -> 2B), [79:48]=tensor_dim0, [111:80]=tensor_dim1,
//          [127:112]=tile_dim0, [143:128]=tile_dim1, [207:160]=tensor_dim0_stride
static __device__ __forceinline__ void tdm_load_2d(unsigned ldsOff, const _Float16* gptr,
                                                   unsigned tile0, unsigned tile1,
                                                   unsigned dim0, unsigned dim1,
                                                   unsigned stride0)
{
  unsigned long long ga = (unsigned long long)(uintptr_t)gptr;
  u32x4 g0;
  g0[0] = 0x1u;                                   // count=1, user descriptor
  g0[1] = ldsOff;                                 // LDS byte address
  g0[2] = (unsigned)(ga & 0xffffffffu);           // global addr [31:0]
  g0[3] = (unsigned)((ga >> 32) & 0x01ffffffu) | (2u << 30);  // addr[56:32] | type=2
  i32x8 g1;
  g1[0] = 0x00010000;                             // workgroup_mask=0, data_size=1 (2 bytes)
  g1[1] = (int)(dim0 << 16);                      // tensor_dim0[15:0]
  g1[2] = (int)((dim0 >> 16) | (dim1 << 16));     // tensor_dim0[31:16] | tensor_dim1[15:0]
  g1[3] = (int)((dim1 >> 16) | (tile0 << 16));    // tensor_dim1[31:16] | tile_dim0
  g1[4] = (int)(tile1 & 0xffffu);                 // tile_dim1 (tile_dim2 = 0 -> 2D)
  g1[5] = (int)stride0;                           // tensor_dim0_stride[31:0]
  g1[6] = 0;
  g1[7] = 0;
  i32x4 z4 = {0, 0, 0, 0};
  i32x8 z8 = {0, 0, 0, 0, 0, 0, 0, 0};
  __builtin_amdgcn_tensor_load_to_lds(g0, g1, z4, z4, z8, 0);
}
#endif

// ---------------------------------------------------------------------------
// Conv2d(3->8, k=1, pad=1) + bias for q/k/v, packed to f16.
// q,k stored row-major [512][row][col]; v stored TRANSPOSED [512][col][row]
// so the P*V B-fragments are contiguous later.
// ---------------------------------------------------------------------------
__global__ __launch_bounds__(256)
void conv_pack_kernel(const float* __restrict__ qin, const float* __restrict__ kin,
                      const float* __restrict__ vin, const float* __restrict__ w,
                      const float* __restrict__ bias,
                      _Float16* __restrict__ qh, _Float16* __restrict__ kh,
                      _Float16* __restrict__ vt)
{
  int idx = blockIdx.x * 256 + threadIdx.x;        // 64*256*256 threads
  int x = idx & 255;
  int y = (idx >> 8) & 255;
  int b = idx >> 16;

  float iq[3] = {0.f,0.f,0.f}, ik[3] = {0.f,0.f,0.f}, iv[3] = {0.f,0.f,0.f};
  if (x >= 1 && x <= 254 && y >= 1 && y <= 254) {
    int iy = y - 1, ix = x - 1;
    #pragma unroll
    for (int i = 0; i < 3; ++i) {
      int off = ((b * 3 + i) * 254 + iy) * 254 + ix;
      iq[i] = qin[off]; ik[i] = kin[off]; iv[i] = vin[off];
    }
  }
  int obase = (b << 19) + (y << 8) + x;            // row-major
  int tbase = (b << 19) + (x << 8) + y;            // transposed (v)
  #pragma unroll
  for (int c = 0; c < 8; ++c) {
    float w0 = w[c*3+0], w1 = w[c*3+1], w2 = w[c*3+2], bb = bias[c];
    qh[obase + (c << 16)] = (_Float16)(bb + w0*iq[0] + w1*iq[1] + w2*iq[2]);
    kh[obase + (c << 16)] = (_Float16)(bb + w0*ik[0] + w1*ik[1] + w2*ik[2]);
    vt[tbase + (c << 16)] = (_Float16)(bb + w0*iv[0] + w1*iv[1] + w2*iv[2]);
  }
}

// ---------------------------------------------------------------------------
// Flash attention, one 512-thread block (16 wave32) per slice.
// Q (whole slice) + double-buffered K / V^T blocks in LDS (208 KB of the
// WGP's 320 KB). The Tensor Data Mover fetches block j+1 while the WMMAs
// consume block j; s_wait_tensorcnt<=2 exploits in-order TDM completion.
// ---------------------------------------------------------------------------
__global__ __launch_bounds__(512)
void flash_attn_kernel(const _Float16* __restrict__ qh, const _Float16* __restrict__ kh,
                       const _Float16* __restrict__ vth, const float* __restrict__ mask,
                       float* __restrict__ out)
{
  __shared__ alignas(32) _Float16 Qs[256 * 256];      // full Q slice        (128 KB)
  __shared__ alignas(32) _Float16 Ks[2][32 * 256];    // K rows, dbl-buffer  (32 KB)
  __shared__ alignas(32) _Float16 Vt[2][256 * 32];    // V^T,   dbl-buffer   (32 KB)
  __shared__ alignas(32) _Float16 Ps[16][16 * 32];    // per-wave P scratch  (16 KB)

  const int lane  = threadIdx.x & 31;
  const int wave  = threadIdx.x >> 5;
  const int slice = blockIdx.x;
  const int rowBlk = wave << 4;                     // 16 rows per wave

  const int nlow     = lane & 15;
  const int hiA      = (lane >> 4) << 3;            // A-frag K chunk select: 0 / 8
  const int hiB      = (lane >> 4) << 4;            // B-frag K half select:  0 / 16
  const int mrowHalf = (lane >> 4) << 3;            // C layout: lanes>=16 hold M+8

  const _Float16* Q   = qh  + (size_t)slice * (SEQ * DHEAD);
  const _Float16* K   = kh  + (size_t)slice * (SEQ * DHEAD);
  const _Float16* Vtg = vth + (size_t)slice * (SEQ * DHEAD);

#if HAVE_TDM
  const unsigned qsOff  = (unsigned)(uintptr_t)&Qs[0];
  const unsigned ksOff0 = (unsigned)(uintptr_t)&Ks[0][0];
  const unsigned ksOff1 = (unsigned)(uintptr_t)&Ks[1][0];
  const unsigned vtOff0 = (unsigned)(uintptr_t)&Vt[0][0];
  const unsigned vtOff1 = (unsigned)(uintptr_t)&Vt[1][0];
  if (wave == 0) {
    tdm_load_2d(qsOff,  Q,   256, 256, 256, 256, 256);  // whole Q slice, once
    tdm_load_2d(ksOff0, K,   256,  32, 256, 256, 256);  // K block 0
    tdm_load_2d(vtOff0, Vtg,  32, 256, 256, 256, 256);  // V^T block 0
  }
#else
  #pragma unroll
  for (int it = 0; it < 16; ++it) {
    int o = (it * 512 + threadIdx.x) * 8;
    *(v8h*)(Qs + o) = *(const v8h*)(Q + o);
  }
#endif

  const v8f zero8 = {0.f,0.f,0.f,0.f,0.f,0.f,0.f,0.f};
  v8f acc[16];                                      // O: 16 rows x 256 cols, f32
  #pragma unroll
  for (int t = 0; t < 16; ++t) acc[t] = zero8;
  float mrow[8], lrow[8];
  #pragma unroll
  for (int v = 0; v < 8; ++v) { mrow[v] = -3.0e38f; lrow[v] = 0.f; }

  _Float16* P = &Ps[wave][0];
  const _Float16* qrowLds = Qs + (rowBlk + nlow) * DHEAD;

  for (int j = 0; j < 8; ++j) {                     // 32 keys per iteration
    // all waves done reading the buffer that is about to be overwritten
    if (j) __syncthreads();

#if HAVE_TDM
    if (wave == 0) {
      if (j < 7) {                                  // issue j+1 ahead of compute
        unsigned ksN = ((j + 1) & 1) ? ksOff1 : ksOff0;
        unsigned vtN = ((j + 1) & 1) ? vtOff1 : vtOff0;
        tdm_load_2d(ksN, K + (j + 1) * 32 * DHEAD, 256,  32, 256, 256, 256);
        tdm_load_2d(vtN, Vtg + (j + 1) * 32,        32, 256, 256, 256, 256);
        __builtin_amdgcn_s_wait_tensorcnt(2);       // block j (and Q) complete
      } else {
        __builtin_amdgcn_s_wait_tensorcnt(0);
      }
    }
#else
    {
      const _Float16* Kg = K + j * 32 * DHEAD;
      _Float16* ksB = &Ks[j & 1][0];
      _Float16* vtB = &Vt[j & 1][0];
      #pragma unroll
      for (int it = 0; it < 2; ++it) {
        int o = (it * 512 + threadIdx.x) * 8;
        *(v8h*)(ksB + o) = *(const v8h*)(Kg + o);
      }
      int c = threadIdx.x >> 1, koff = (threadIdx.x & 1) << 4;
      *(v16h*)(vtB + c * 32 + koff) = *(const v16h*)(Vtg + c * 256 + j * 32 + koff);
    }
#endif
    if (j < 6) {                                    // global_prefetch_b8: block j+2 to L2
      __builtin_prefetch(K + (j + 2) * 32 * DHEAD + (threadIdx.x << 4), 0, 0);
      __builtin_prefetch(Vtg + j * 32 + 64 + ((threadIdx.x >> 1) << 8), 0, 0);
    }
    __syncthreads();                                // publish block j

    const _Float16* Ksb = &Ks[j & 1][0];
    const _Float16* Vtb = &Vt[j & 1][0];

    // ---- S = Q * K^T : two 16x16 tiles, K-depth 256 (A and B from LDS) ----
    v8f s0 = zero8, s1 = zero8;
    #pragma unroll
    for (int kk = 0; kk < 8; ++kk) {
      v8h alo = *(const v8h*)(qrowLds + kk * 32 + hiB);
      v8h ahi = *(const v8h*)(qrowLds + kk * 32 + 8 + hiB);
      v16h af = cat8(alo, ahi);
      v16h b0 = *(const v16h*)(Ksb + nlow * 256 + kk * 32 + hiB);
      s0 = __builtin_amdgcn_wmma_f32_16x16x32_f16(false, af, false, b0,
                                                  (short)0, s0, false, false);
      v16h b1 = *(const v16h*)(Ksb + (16 + nlow) * 256 + kk * 32 + hiB);
      s1 = __builtin_amdgcn_wmma_f32_16x16x32_f16(false, af, false, b1,
                                                  (short)0, s1, false, false);
    }
    float mk0 = mask[j * 32 + nlow];                // scale = 1/sqrt(1) = 1
    float mk1 = mask[j * 32 + 16 + nlow];

    // ---- online softmax (row = fixed VGPR idx, 16 lanes per half-wave) ----
    float p0a[8], p1a[8];
    #pragma unroll
    for (int v = 0; v < 8; ++v) {
      float a0 = s0[v] + mk0;
      float a1 = s1[v] + mk1;
      float mx = fmaxf(a0, a1);
      mx = fmaxf(mx, __shfl_xor(mx, 1));
      mx = fmaxf(mx, __shfl_xor(mx, 2));
      mx = fmaxf(mx, __shfl_xor(mx, 4));
      mx = fmaxf(mx, __shfl_xor(mx, 8));
      float newm  = fmaxf(mrow[v], mx);
      float alpha = __expf(mrow[v] - newm);
      mrow[v] = newm;
      float p0 = __expf(a0 - newm);
      float p1 = __expf(a1 - newm);
      p0a[v] = p0; p1a[v] = p1;
      float rs = p0 + p1;
      rs += __shfl_xor(rs, 1);
      rs += __shfl_xor(rs, 2);
      rs += __shfl_xor(rs, 4);
      rs += __shfl_xor(rs, 8);
      lrow[v] = lrow[v] * alpha + rs;
      #pragma unroll
      for (int t = 0; t < 16; ++t) acc[t][v] *= alpha;
    }

    // ---- reshape P (C layout -> A fragment) through per-wave LDS ----
    #pragma unroll
    for (int v = 0; v < 8; ++v) {
      int row = v + mrowHalf;
      P[row * 32 + nlow]      = (_Float16)p0a[v];
      P[row * 32 + 16 + nlow] = (_Float16)p1a[v];
    }
    v16h pa;
    {
      const _Float16* prow = P + nlow * 32;
      v8h lo = *(const v8h*)(prow + hiB);
      v8h hi = *(const v8h*)(prow + 8 + hiB);
      pa = cat8(lo, hi);
    }

    // ---- O += P * V ----
    #pragma unroll
    for (int t = 0; t < 16; ++t) {
      v16h bv = *(const v16h*)(Vtb + (t * 16 + nlow) * 32 + hiB);
      acc[t] = __builtin_amdgcn_wmma_f32_16x16x32_f16(false, pa, false, bv,
                                                      (short)0, acc[t], false, false);
    }
  }

  // ---- normalize by row sums and store fp32 ----
  float inv[8];
  #pragma unroll
  for (int v = 0; v < 8; ++v) inv[v] = 1.f / lrow[v];
  float* O = out + (size_t)slice * (SEQ * DHEAD);
  #pragma unroll
  for (int t = 0; t < 16; ++t) {
    #pragma unroll
    for (int v = 0; v < 8; ++v) {
      O[(rowBlk + v + mrowHalf) * 256 + t * 16 + nlow] = acc[t][v] * inv[v];
    }
  }
}

// ---------------------------------------------------------------------------
extern "C" void kernel_launch(void* const* d_in, const int* in_sizes, int n_in,
                              void* d_out, int out_size, void* d_ws, size_t ws_size,
                              hipStream_t stream) {
  const float* qin  = (const float*)d_in[0];
  const float* kin  = (const float*)d_in[1];
  const float* vin  = (const float*)d_in[2];
  const float* mask = (const float*)d_in[3];
  const float* w    = (const float*)d_in[4];
  const float* bias = (const float*)d_in[5];
  float* out = (float*)d_out;

  const size_t sliceElems = (size_t)NSLICE * SEQ * DHEAD;  // 33,554,432 halves / tensor
  _Float16* qh = (_Float16*)d_ws;                          // needs 3*64 MB of ws
  _Float16* kh = qh + sliceElems;
  _Float16* vt = kh + sliceElems;

  // 64*256*256 output pixels / 256 threads
  conv_pack_kernel<<<16384, 256, 0, stream>>>(qin, kin, vin, w, bias, qh, kh, vt);
  // one block per (batch, channel) slice
  flash_attn_kernel<<<NSLICE, 512, 0, stream>>>(qh, kh, vt, mask, out);
}
